// MLSTMCharLMLayer_40982577938638
// MI455X (gfx1250) — matrix operand order repaired and checked by reading, not compiled, not run on hardware
//
#include <hip/hip_runtime.h>

// ---------------------------------------------------------------- constants
#define NVOCAB 256
#define NEMBD  64
#define NHID   4096
#define N4H    16384
#define NB     64
#define NT     256

typedef __bf16 bf16_t;
typedef __attribute__((ext_vector_type(16))) __bf16 v16bf;
typedef __attribute__((ext_vector_type(8)))  __bf16 v8bf;
typedef __attribute__((ext_vector_type(8)))  float  v8f;

// LDS tile geometry: each staged row = 32 bf16 of K (64 B data) padded to 80 B
// to spread LDS banks. W tile = 128 rows (output columns), A tile = 16*ROWT rows.
constexpr int LROW  = 80;
constexpr int WROWS = 128;
constexpr int WBUF  = WROWS * LROW;          // 10240 B per W buffer

// ---------------------------------------------------------------- async copy
// CDNA5 async global->LDS DMA (no VGPR staging, tracked by ASYNCcnt).
// ISA pseudocode applies INST_OFFSET to BOTH the LDS and global address, so a
// single (lds,vaddr) operand pair covers a 32B chunk with offset:0 / offset:16.
__device__ __forceinline__ void async_copy32(const void* gsrc, void* ldst) {
  const unsigned lds = (unsigned)(uintptr_t)ldst;               // flat low 32b == DS addr
  const unsigned long long ga = (unsigned long long)(uintptr_t)gsrc;
  asm volatile(
      "global_load_async_to_lds_b128 %0, %1, off\n\t"
      "global_load_async_to_lds_b128 %0, %1, off offset:16"
      :: "v"(lds), "v"(ga) : "memory");
}

__device__ __forceinline__ void wait_async0() {
  asm volatile("s_wait_asynccnt 0" ::: "memory");
}

// ---------------------------------------------------------------- fragments
__device__ __forceinline__ v16bf make16(v8bf lo, v8bf hi) {
  v16bf r;
#pragma unroll
  for (int i = 0; i < 8; ++i) { r[i] = lo[i]; r[i + 8] = hi[i]; }
  return r;
}

// B fragment (K=32 x N=16): lane j holds column (j&15), K = (j>>4)*16 .. +15
__device__ __forceinline__ v16bf load_bfrag(const char* wbuf, int wave, int lane) {
  const int r = wave * 16 + (lane & 15);
  const char* p = wbuf + r * LROW + ((lane >> 4) * 32);
  return make16(*(const v8bf*)p, *(const v8bf*)(p + 16));
}

// A fragment (M=16 x K=32): lane j holds row (j&15);
// lanes<16: K 0..15, lanes>=16: K 16..31
__device__ __forceinline__ v16bf load_afrag(const char* abuf, int rt, int lane) {
  const int r = rt * 16 + (lane & 15);
  const char* p = abuf + r * LROW + ((lane >> 4) * 32);
  return make16(*(const v8bf*)p, *(const v8bf*)(p + 16));
}

// ---------------------------------------------------------------- LDS fills
// 256 threads fill 128 rows x 64B (thread: one 32B chunk = two async b128)
__device__ __forceinline__ void fill_tileW(char* dst, const bf16_t* __restrict__ Wt,
                                           int K, int nBase, int k0, int tid) {
  const int row = tid >> 1, half = tid & 1;
  async_copy32(Wt + (size_t)(nBase + row) * K + k0 + half * 16,
               dst + row * LROW + half * 32);
}

__device__ __forceinline__ void fill_tileA(char* dst, const bf16_t* __restrict__ A,
                                           int lda, int mBase, int k0, int tid, int rows) {
  if (tid < rows * 2) {
    const int row = tid >> 1, half = tid & 1;
    async_copy32(A + (size_t)(mBase + row) * lda + k0 + half * 16,
                 dst + row * LROW + half * 32);
  }
}

// ---------------------------------------------------------------- WMMA K-loop
// C[16*ROWT x 128 tile] += A[16*ROWT x K] * Wt^T  (Wt stored [N,K] row-major)
// 8 waves: wave = 16-column tile; double-buffered LDS filled by async DMA;
// bf16 in, f32 accumulate.
template <int ROWT>
__device__ __forceinline__ void gemm_pipeline(const bf16_t* __restrict__ A, int lda, int mBase,
                                              const bf16_t* __restrict__ Wt, int K, int nBase,
                                              char* ldsW, char* ldsA, v8f* acc) {
  constexpr int AROWS = ROWT * 16;
  constexpr int ABUF  = AROWS * LROW;
  const int tid  = threadIdx.x;
  const int wave = tid >> 5;
  const int lane = tid & 31;

  fill_tileW(ldsW, Wt, K, nBase, 0, tid);
  fill_tileA(ldsA, A, lda, mBase, 0, tid, AROWS);
  wait_async0();
  __syncthreads();

  for (int k0 = 0; k0 < K; k0 += 32) {
    const int cur = (k0 >> 5) & 1;
    const int nxt = cur ^ 1;
    if (k0 + 64 < K)  // pull K-tile(+2) toward L2 (global_prefetch_b8)
      __builtin_prefetch(Wt + (size_t)(nBase + (tid >> 1)) * K + k0 + 64, 0, 1);
    if (k0 + 32 < K) {
      fill_tileW(ldsW + nxt * WBUF, Wt, K, nBase, k0 + 32, tid);
      fill_tileA(ldsA + nxt * ABUF, A, lda, mBase, k0 + 32, tid, AROWS);
    }
    const char* wbuf = ldsW + cur * WBUF;
    const char* abuf = ldsA + cur * ABUF;
    v16bf bfrag = load_bfrag(wbuf, wave, lane);
#pragma unroll
    for (int rt = 0; rt < ROWT; ++rt) {
      v16bf afrag = load_afrag(abuf, rt, lane);
      acc[rt] = __builtin_amdgcn_wmma_f32_16x16x32_bf16(
          false, afrag, false, bfrag, (short)0, acc[rt], false, false);
    }
    wait_async0();    // next-buffer DMA done (overlapped with the MMAs above)
    __syncthreads();
  }
}

// ---------------------------------------------------------------- prologue
// per-output-column inv-norm: inv[n] = g[n] / sqrt(sum_k w[k][n]^2 + 1e-12)
__global__ void __launch_bounds__(256) k_colnorm(const float* __restrict__ w,
                                                 const float* __restrict__ g,
                                                 float* __restrict__ inv, int K, int N) {
  __shared__ float red[256];
  const int base = blockIdx.x * 64;
  const int cc = threadIdx.x & 63;
  const int kk = threadIdx.x >> 6;
  const int col = base + cc;
  float s = 0.f;
  for (int k = kk; k < K; k += 4) {
    float x = w[(size_t)k * N + col];
    s += x * x;
  }
  red[threadIdx.x] = s;
  __syncthreads();
  if (threadIdx.x < 64) {
    float t = red[cc] + red[cc + 64] + red[cc + 128] + red[cc + 192];
    inv[col] = g[col] * rsqrtf(t + 1e-12f);
  }
}

// transpose [K,N] f32 -> [N,K] bf16, scaled by inv[n] (inv==nullptr => 1.0)
__global__ void __launch_bounds__(256) k_castT(const float* __restrict__ w,
                                               const float* __restrict__ inv,
                                               bf16_t* __restrict__ wt, int K, int N) {
  __shared__ float tile[32][33];
  const int nb = blockIdx.x * 32, kb = blockIdx.y * 32;
  const int tx = threadIdx.x & 31, ty = threadIdx.x >> 5;
#pragma unroll
  for (int j = 0; j < 4; ++j)
    tile[ty + j * 8][tx] = w[(size_t)(kb + ty + j * 8) * N + nb + tx];
  __syncthreads();
#pragma unroll
  for (int j = 0; j < 4; ++j) {
    const int n = nb + ty + j * 8;
    const float sc = inv ? inv[n] : 1.0f;
    wt[(size_t)n * K + kb + tx] = (bf16_t)(tile[tx][ty + j * 8] * sc);
  }
}

// xs[t][b][e] = bf16(embd[X[b][t]][e])
__global__ void __launch_bounds__(256) k_embed(const int* __restrict__ X,
                                               const float* __restrict__ E,
                                               bf16_t* __restrict__ xs) {
  const int i = blockIdx.x * blockDim.x + threadIdx.x;  // T*B*64
  const int e = i & 63;
  const int b = (i >> 6) & 63;
  const int t = i >> 12;
  const int tok = X[b * NT + t];
  xs[i] = (bf16_t)E[tok * NEMBD + e];
}

__global__ void __launch_bounds__(256) k_init(const float* __restrict__ S,
                                              float* __restrict__ c, float* __restrict__ h,
                                              bf16_t* __restrict__ h_bf) {
  const int i = blockIdx.x * blockDim.x + threadIdx.x;  // NB*NHID
  const float cv = S[i];
  const float hv = S[NB * NHID + i];
  c[i] = cv;
  h[i] = hv;
  h_bf[i] = (bf16_t)hv;
}

// ---------------------------------------------------------------- recurrent
// mgate = (x@wmx_n) * (h@wmh_n), bf16 out
__global__ void __launch_bounds__(256) k_mgate(const bf16_t* __restrict__ h_bf,
                                               const bf16_t* __restrict__ wmh_t,
                                               const bf16_t* __restrict__ x_bf,
                                               const bf16_t* __restrict__ wmx_t,
                                               bf16_t* __restrict__ mg) {
  extern __shared__ char smem[];
  char* ldsW = smem;
  char* ldsA = smem + 2 * WBUF;
  const int nBase = blockIdx.x * 128;
  v8f acc1[4] = {};
  v8f acc2[4] = {};
  gemm_pipeline<4>(h_bf, NHID, 0, wmh_t, NHID, nBase, ldsW, ldsA, acc1);
  gemm_pipeline<4>(x_bf, NEMBD, 0, wmx_t, NEMBD, nBase, ldsW, ldsA, acc2);
  const int wave = threadIdx.x >> 5, lane = threadIdx.x & 31;
  const int col = nBase + wave * 16 + (lane & 15);
  const int rb = (lane >> 4) * 8;
#pragma unroll
  for (int rt = 0; rt < 4; ++rt)
#pragma unroll
    for (int v = 0; v < 8; ++v) {
      const int m = rt * 16 + rb + v;
      mg[(size_t)m * NHID + col] = (bf16_t)(acc1[rt][v] * acc2[rt][v]);
    }
}

// z = mgate@wh_n + x@wx_n + b, f32 out
__global__ void __launch_bounds__(256) k_zgemm(const bf16_t* __restrict__ mg_bf,
                                               const bf16_t* __restrict__ wh_t,
                                               const bf16_t* __restrict__ x_bf,
                                               const bf16_t* __restrict__ wx_t,
                                               const float* __restrict__ bias,
                                               float* __restrict__ z) {
  extern __shared__ char smem[];
  char* ldsW = smem;
  char* ldsA = smem + 2 * WBUF;
  const int nBase = blockIdx.x * 128;
  v8f acc[4] = {};
  gemm_pipeline<4>(mg_bf, NHID, 0, wh_t, NHID, nBase, ldsW, ldsA, acc);
  gemm_pipeline<4>(x_bf, NEMBD, 0, wx_t, NEMBD, nBase, ldsW, ldsA, acc);
  const int wave = threadIdx.x >> 5, lane = threadIdx.x & 31;
  const int col = nBase + wave * 16 + (lane & 15);
  const int rb = (lane >> 4) * 8;
  const float bc = bias[col];
#pragma unroll
  for (int rt = 0; rt < 4; ++rt)
#pragma unroll
    for (int v = 0; v < 8; ++v) {
      const int m = rt * 16 + rb + v;
      z[(size_t)m * N4H + col] = acc[rt][v] + bc;
    }
}

// gates + masked state update; writes cells[t], c, h (f32) and h (bf16) copies
__global__ void __launch_bounds__(256) k_point(const float* __restrict__ z,
                                               const float* __restrict__ Mm, int t,
                                               float* __restrict__ c, float* __restrict__ h,
                                               bf16_t* __restrict__ h_bf,
                                               bf16_t* __restrict__ hs_bf,
                                               float* __restrict__ cells) {
  const int i = blockIdx.x * blockDim.x + threadIdx.x;  // NB*NHID
  const int b = i >> 12;
  const int n = i & 4095;
  const float* zb = z + (size_t)b * N4H;
  const float zi = zb[n];
  const float zf = zb[n + NHID];
  const float zo = zb[n + 2 * NHID];
  const float zu = zb[n + 3 * NHID];
  const float ig = 1.f / (1.f + __expf(-zi));
  const float fg = 1.f / (1.f + __expf(-zf));
  const float og = 1.f / (1.f + __expf(-zo));
  const float ug = tanhf(zu);
  const float c0 = c[i], h0 = h[i];
  const float ct = fg * c0 + ig * ug;
  const float ht = og * tanhf(ct);
  const float m = Mm[b * NT + t];
  const float cn = ct * m + c0 * (1.f - m);
  const float hn = ht * m + h0 * (1.f - m);
  c[i] = cn;
  h[i] = hn;
  h_bf[i] = (bf16_t)hn;
  hs_bf[i] = (bf16_t)hn;
  cells[i] = cn;
}

// ---------------------------------------------------------------- epilogue
// logits = hs @ out_w + out_b  (M=16384, N=256, K=4096)
__global__ void __launch_bounds__(256) k_logits(const bf16_t* __restrict__ hs_bf,
                                                const bf16_t* __restrict__ owt,
                                                const float* __restrict__ ob,
                                                float* __restrict__ out) {
  extern __shared__ char smem[];
  char* ldsW = smem;
  char* ldsA = smem + 2 * WBUF;
  const int nBase = blockIdx.x * 128;
  const int mBase = blockIdx.y * 128;
  v8f acc[8] = {};
  gemm_pipeline<8>(hs_bf, NHID, mBase, owt, NHID, nBase, ldsW, ldsA, acc);
  const int wave = threadIdx.x >> 5, lane = threadIdx.x & 31;
  const int col = nBase + wave * 16 + (lane & 15);
  const int rb = (lane >> 4) * 8;
  const float bc = ob[col];
#pragma unroll
  for (int rt = 0; rt < 8; ++rt)
#pragma unroll
    for (int v = 0; v < 8; ++v) {
      const int m = mBase + rt * 16 + rb + v;
      out[(size_t)m * NVOCAB + col] = acc[rt][v] + bc;
    }
}

__global__ void __launch_bounds__(256) k_states(const float* __restrict__ c,
                                                const float* __restrict__ h,
                                                float* __restrict__ out) {
  const int i = blockIdx.x * blockDim.x + threadIdx.x;  // NB*NHID
  out[i] = c[i];
  out[NB * NHID + i] = h[i];
}

// ---------------------------------------------------------------- host
extern "C" void kernel_launch(void* const* d_in, const int* in_sizes, int n_in,
                              void* d_out, int out_size, void* d_ws, size_t ws_size,
                              hipStream_t stream) {
  (void)in_sizes; (void)n_in; (void)out_size; (void)ws_size;
  const int*   X    = (const int*)  d_in[0];
  const float* Mm   = (const float*)d_in[1];
  const float* S    = (const float*)d_in[2];
  const float* embd = (const float*)d_in[3];
  const float* wx   = (const float*)d_in[4];
  const float* wh   = (const float*)d_in[5];
  const float* wmx  = (const float*)d_in[6];
  const float* wmh  = (const float*)d_in[7];
  const float* bias = (const float*)d_in[8];
  const float* gx   = (const float*)d_in[9];
  const float* gh   = (const float*)d_in[10];
  const float* gmx  = (const float*)d_in[11];
  const float* gmh  = (const float*)d_in[12];
  const float* outw = (const float*)d_in[13];
  const float* outb = (const float*)d_in[14];

  char* wsp = (char*)d_ws;
  auto carve = [&](size_t bytes) -> char* {
    char* p = wsp;
    wsp += (bytes + 255) & ~((size_t)255);
    return p;
  };
  bf16_t* wx_t  = (bf16_t*)carve((size_t)N4H * NEMBD * 2);   //   2 MB
  bf16_t* wh_t  = (bf16_t*)carve((size_t)N4H * NHID * 2);    // 128 MB (L2-resident)
  bf16_t* wmx_t = (bf16_t*)carve((size_t)NHID * NEMBD * 2);
  bf16_t* wmh_t = (bf16_t*)carve((size_t)NHID * NHID * 2);   //  32 MB (L2-resident)
  bf16_t* owt   = (bf16_t*)carve((size_t)NVOCAB * NHID * 2);
  float*  inv   = (float*) carve((size_t)N4H * 4);
  bf16_t* xs    = (bf16_t*)carve((size_t)NT * NB * NEMBD * 2);
  bf16_t* mg    = (bf16_t*)carve((size_t)NB * NHID * 2);
  bf16_t* h_bf  = (bf16_t*)carve((size_t)NB * NHID * 2);
  float*  zbuf  = (float*) carve((size_t)NB * N4H * 4);
  float*  cbuf  = (float*) carve((size_t)NB * NHID * 4);
  float*  hbuf  = (float*) carve((size_t)NB * NHID * 4);
  bf16_t* hs    = (bf16_t*)carve((size_t)NT * NB * NHID * 2);

  float* cells  = (float*)d_out;
  float* states = cells + (size_t)NT * NB * NHID;
  float* logits = states + (size_t)2 * NB * NHID;

  // --- weight norm + transpose/cast to bf16 [N,K]
  k_colnorm<<<N4H / 64, 256, 0, stream>>>(wx, gx, inv, NEMBD, N4H);
  k_castT<<<dim3(N4H / 32, NEMBD / 32), 256, 0, stream>>>(wx, inv, wx_t, NEMBD, N4H);
  k_colnorm<<<N4H / 64, 256, 0, stream>>>(wh, gh, inv, NHID, N4H);
  k_castT<<<dim3(N4H / 32, NHID / 32), 256, 0, stream>>>(wh, inv, wh_t, NHID, N4H);
  k_colnorm<<<NHID / 64, 256, 0, stream>>>(wmx, gmx, inv, NEMBD, NHID);
  k_castT<<<dim3(NHID / 32, NEMBD / 32), 256, 0, stream>>>(wmx, inv, wmx_t, NEMBD, NHID);
  k_colnorm<<<NHID / 64, 256, 0, stream>>>(wmh, gmh, inv, NHID, NHID);
  k_castT<<<dim3(NHID / 32, NHID / 32), 256, 0, stream>>>(wmh, inv, wmh_t, NHID, NHID);
  k_castT<<<dim3(NVOCAB / 32, NHID / 32), 256, 0, stream>>>(outw, nullptr, owt, NHID, NVOCAB);

  k_embed<<<(NT * NB * NEMBD) / 256, 256, 0, stream>>>(X, embd, xs);
  k_init<<<(NB * NHID) / 256, 256, 0, stream>>>(S, cbuf, hbuf, h_bf);

  constexpr size_t SMEM_G = 2 * WBUF + 2 * (64 * LROW);    // 30720 B
  constexpr size_t SMEM_L = 2 * WBUF + 2 * (128 * LROW);   // 40960 B

  // --- sequential recurrence (bf16 weights served out of L2)
  for (int t = 0; t < NT; ++t) {
    const bf16_t* xt = xs + (size_t)t * NB * NEMBD;
    k_mgate<<<NHID / 128, 256, SMEM_G, stream>>>(h_bf, wmh_t, xt, wmx_t, mg);
    k_zgemm<<<N4H / 128, 256, SMEM_G, stream>>>(mg, wh_t, xt, wx_t, bias, zbuf);
    k_point<<<(NB * NHID) / 256, 256, 0, stream>>>(zbuf, Mm, t, cbuf, hbuf, h_bf,
                                                   hs + (size_t)t * NB * NHID,
                                                   cells + (size_t)t * NB * NHID);
  }

  // --- logits + final states
  k_logits<<<dim3(NVOCAB / 128, (NT * NB) / 128), 256, SMEM_L, stream>>>(hs, owt, outb, logits);
  k_states<<<(NB * NHID) / 256, 256, 0, stream>>>(cbuf, hbuf, states);
}
